// DCNV3_Yolo11_18640158064724
// MI455X (gfx1250) — matrix operand order repaired and checked by reading, not compiled, not run on hardware
//
#include <hip/hip_runtime.h>
#include <hip/hip_bf16.h>

typedef __attribute__((ext_vector_type(16))) _Float16 v16h;
typedef __attribute__((ext_vector_type(8)))  _Float16 v8h;
typedef __attribute__((ext_vector_type(8)))  float    v8f;

#define NB     2
#define CCH    256
#define HWD    64
#define MROWS  (NB*HWD*HWD)     // 8192
#define HP     66               // padded spatial
#define GRP    8
#define PP     9
#define NOFF   144              // G*P*2
#define NMASK  72               // G*P
#define NOM    216              // 144+72
#define NOMP   224              // padded: 7 blocks of 32

__device__ __forceinline__ float silu_f(float v) { return v / (1.f + expf(-v)); }

// A 16x32 f16 fragment (ISA layout): half 0 -> K {0..7,16..23}, half 1 -> K {8..15,24..31}
__device__ __forceinline__ v16h load_afrag(const _Float16* __restrict__ row, int k0, int half) {
    v8h a0 = *(const v8h*)(row + k0 + half * 8);
    v8h a1 = *(const v8h*)(row + k0 + 16 + half * 8);
    return __builtin_shufflevector(a0, a1, 0,1,2,3,4,5,6,7,8,9,10,11,12,13,14,15);
}
// B 32x16 fragment: lane l holds N=l; half 0 -> K {0..7,16..23}, half 1 -> K {8..15,24..31}
__device__ __forceinline__ v16h load_bfrag(const _Float16* __restrict__ row, int k0, int half) {
    v8h b0 = *(const v8h*)(row + k0 + half * 8);
    v8h b1 = *(const v8h*)(row + k0 + 16 + half * 8);
    return __builtin_shufflevector(b0, b1, 0,1,2,3,4,5,6,7,8,9,10,11,12,13,14,15);
}
__device__ __forceinline__ v8f wmma_f16(v16h a, v16h b, v8f c) {
    return __builtin_amdgcn_wmma_f32_16x16x32_f16(false, a, false, b, (short)0, c, false, false);
}

// ---------------------------------------------------------------------------
// Weight prep: convert/transposed-pack all GEMM B operands to f16 [ocol][K]
// ---------------------------------------------------------------------------
__global__ __launch_bounds__(256) void k_prep_weights(
    const float* __restrict__ conv1_w,   // (256,256) already [o][c]
    const float* __restrict__ inproj_w,  // (256,256) used as t@W -> need [o][c]=W[c][o]
    const float* __restrict__ off_w,     // (256,144)
    const float* __restrict__ mask_w,    // (256,72)
    const float* __restrict__ outproj_w, // (256,256)
    const float* __restrict__ off_b, const float* __restrict__ mask_b,
    _Float16* __restrict__ wt_c1, _Float16* __restrict__ wt_in,
    _Float16* __restrict__ wt_om, _Float16* __restrict__ wt_out,
    float* __restrict__ bias_om)
{
    int o = blockIdx.x;      // 0..255
    int c = threadIdx.x;     // 0..255
    wt_c1 [o*CCH + c] = (_Float16)conv1_w [o*CCH + c];
    wt_in [o*CCH + c] = (_Float16)inproj_w[c*CCH + o];
    wt_out[o*CCH + c] = (_Float16)outproj_w[c*CCH + o];
    if (o < NOMP) {
        float v = 0.f;
        if (o < NOFF)      v = off_w [c*NOFF + o];
        else if (o < NOM)  v = mask_w[c*NMASK + (o - NOFF)];
        wt_om[o*CCH + c] = (_Float16)v;
        if (c == 0) {
            float b = 0.f;
            if (o < NOFF)     b = off_b[o];
            else if (o < NOM) b = mask_b[o - NOFF];
            bias_om[o] = b;
        }
    }
}

// ---------------------------------------------------------------------------
// NCHW f32 -> NHWC f16 (A matrix of GEMM1)
// ---------------------------------------------------------------------------
__global__ __launch_bounds__(256) void k_x_to_nhwc16(
    const float* __restrict__ x, _Float16* __restrict__ a1)
{
    int t = blockIdx.x * blockDim.x + threadIdx.x;   // NCHW linear, coalesced read
    int w = t & 63, h = (t >> 6) & 63, c = (t >> 12) & 255, n = t >> 20;
    int m = (n << 12) | (h << 6) | w;
    a1[(size_t)m * CCH + c] = (_Float16)x[t];
}

__global__ __launch_bounds__(256) void k_zero_f32(float* __restrict__ p, int n)
{
    int t = blockIdx.x * blockDim.x + threadIdx.x;
    if (t < n) p[t] = 0.f;
}

// ---------------------------------------------------------------------------
// WMMA GEMM, one wave = 32x32 output (2x2 tiles of 16x16), K=256 unrolled.
//   MODE 0: conv1   -> BN1(eps 1e-3) + SiLU, write f32 + f16 [m][col]
//   MODE 1: inproj  -> +bias, write f32 into padded NHWC image (66x66)
//   MODE 2: offmask -> +bias, write f32 [m][NOMP]
//   MODE 3: outproj -> +bias, BN2(eps 1e-5) + SiLU, scatter NCHW to d_out
// ---------------------------------------------------------------------------
template<int MODE>
__global__ __launch_bounds__(256) void k_gemm(
    const _Float16* __restrict__ A, const _Float16* __restrict__ Wt,
    int nblkN, int Kdim,
    const float* __restrict__ q0, const float* __restrict__ q1,
    const float* __restrict__ q2, const float* __restrict__ q3,
    const float* __restrict__ q4,
    float* __restrict__ o32, _Float16* __restrict__ o16)
{
    int wid  = (blockIdx.x * blockDim.x + threadIdx.x) >> 5;
    int lane = threadIdx.x & 31;
    int half = lane >> 4;          // 0: lanes 0-15, 1: lanes 16-31
    int l    = lane & 15;
    int mblk = wid / nblkN;
    int nblk = wid % nblkN;

    const _Float16* arow0 = A  + (size_t)(mblk * 32 + l) * Kdim;
    const _Float16* arow1 = arow0 + (size_t)16 * Kdim;
    const _Float16* brow0 = Wt + (size_t)(nblk * 32 + l) * Kdim;
    const _Float16* brow1 = brow0 + (size_t)16 * Kdim;

    v8f acc[2][2] = {};
    #pragma unroll
    for (int k0 = 0; k0 < 256; k0 += 32) {
        v16h a0 = load_afrag(arow0, k0, half);
        v16h a1 = load_afrag(arow1, k0, half);
        v16h b0 = load_bfrag(brow0, k0, half);
        v16h b1 = load_bfrag(brow1, k0, half);
        acc[0][0] = wmma_f16(a0, b0, acc[0][0]);
        acc[0][1] = wmma_f16(a0, b1, acc[0][1]);
        acc[1][0] = wmma_f16(a1, b0, acc[1][0]);
        acc[1][1] = wmma_f16(a1, b1, acc[1][1]);
    }

    #pragma unroll
    for (int nt = 0; nt < 2; ++nt) {
        int col = nblk * 32 + nt * 16 + l;
        float scale = 0.f, shift = 0.f, bias = 0.f;
        if (MODE == 0) {
            scale = q0[col] * rsqrtf(q3[col] + 1e-3f);
            shift = q1[col] - q2[col] * scale;
        } else if (MODE == 1 || MODE == 2) {
            bias = q0[col];
        } else {
            bias  = q0[col];
            scale = q1[col] * rsqrtf(q4[col] + 1e-5f);
            shift = q2[col] - q3[col] * scale;
        }
        #pragma unroll
        for (int mt = 0; mt < 2; ++mt) {
            #pragma unroll
            for (int i = 0; i < 8; ++i) {
                int row = mblk * 32 + mt * 16 + i + half * 8;
                float v = acc[mt][nt][i];
                if (MODE == 0) {
                    v = silu_f(v * scale + shift);
                    o32[(size_t)row * CCH + col] = v;
                    o16[(size_t)row * CCH + col] = (_Float16)v;
                } else if (MODE == 1) {
                    int n = row >> 12, hw = row & 4095;
                    int h = hw >> 6, w = hw & 63;
                    size_t idx = (((size_t)n * HP + (h + 1)) * HP + (w + 1)) * CCH + col;
                    o32[idx] = v + bias;
                } else if (MODE == 2) {
                    o32[(size_t)row * NOMP + col] = v + bias;
                } else {
                    int n = row >> 12, hw = row & 4095;
                    float s = silu_f((v + bias) * scale + shift);
                    size_t idx = (((size_t)n * CCH + col) * HWD + (hw >> 6)) * HWD + (hw & 63);
                    o32[idx] = s;          // NCHW output
                }
            }
        }
    }
}

// ---------------------------------------------------------------------------
// Depthwise 3x3 (pad 1) + LayerNorm(C) + exact GELU. One block per pixel.
// ---------------------------------------------------------------------------
__global__ __launch_bounds__(256) void k_dw_ln_gelu(
    const float* __restrict__ y, const float* __restrict__ dw_w,
    const float* __restrict__ dw_b, const float* __restrict__ ln_g,
    const float* __restrict__ ln_b, _Float16* __restrict__ z16)
{
    __shared__ float sred[256];
    int m = blockIdx.x, c = threadIdx.x;
    int w = m & 63, h = (m >> 6) & 63, n = m >> 12;

    float acc = dw_b[c];
    #pragma unroll
    for (int ky = 0; ky < 3; ++ky) {
        int hy = h + ky - 1;
        #pragma unroll
        for (int kx = 0; kx < 3; ++kx) {
            int wx = w + kx - 1;
            if (hy >= 0 && hy < HWD && wx >= 0 && wx < HWD)
                acc += y[(((size_t)n * HWD + hy) * HWD + wx) * CCH + c] *
                       dw_w[c * 9 + ky * 3 + kx];
        }
    }
    sred[c] = acc; __syncthreads();
    for (int s = 128; s > 0; s >>= 1) { if (c < s) sred[c] += sred[c + s]; __syncthreads(); }
    float mu = sred[0] * (1.f / 256.f);
    __syncthreads();
    float d = acc - mu;
    sred[c] = d * d; __syncthreads();
    for (int s = 128; s > 0; s >>= 1) { if (c < s) sred[c] += sred[c + s]; __syncthreads(); }
    float var = sred[0] * (1.f / 256.f);

    float zn = d * rsqrtf(var + 1e-5f) * ln_g[c] + ln_b[c];
    float ge = 0.5f * zn * (1.f + erff(zn * 0.70710678118654752f));
    z16[(size_t)m * CCH + c] = (_Float16)ge;
}

// ---------------------------------------------------------------------------
// Deformable sampling with fused softmax: one wave per (pixel, group),
// lane = channel-in-group. Softmax logits & offsets are wave-uniform loads.
// px = w + ix + offx ; py = h + jy + offy  (coords in 66x66 padded image)
// ---------------------------------------------------------------------------
__global__ __launch_bounds__(256) void k_dcn(
    const float* __restrict__ xpad, const float* __restrict__ om,
    _Float16* __restrict__ s16)
{
    int wid  = (blockIdx.x * blockDim.x + threadIdx.x) >> 5;  // 0..65535
    int lane = threadIdx.x & 31;                              // cg
    int g = wid & 7;
    int m = wid >> 3;
    int w = m & 63, h = (m >> 6) & 63, n = m >> 12;

    const float* omr = om + (size_t)m * NOMP + g * (PP * 2);  // offsets (x,y)*9
    const float* lgr = om + (size_t)m * NOMP + NOFF + g * PP; // mask logits
    int cidx = g * 32 + lane;
    const float* img = xpad + (size_t)n * HP * HP * CCH + cidx;

    // softmax over P=9 (redundant per lane; broadcast loads)
    float lg[PP];
    float mx = -1e30f;
    #pragma unroll
    for (int p = 0; p < PP; ++p) { lg[p] = lgr[p]; mx = fmaxf(mx, lg[p]); }
    float ssum = 0.f;
    #pragma unroll
    for (int p = 0; p < PP; ++p) { lg[p] = expf(lg[p] - mx); ssum += lg[p]; }
    float sinv = 1.f / ssum;

    float acc = 0.f;
    #pragma unroll
    for (int p = 0; p < PP; ++p) {
        int ix = p / 3, jy = p % 3;                 // meshgrid(indexing='ij'): x varies first
        float px = (float)(w + ix) + omr[p * 2 + 0];
        float py = (float)(h + jy) + omr[p * 2 + 1];
        float x0f = floorf(px), y0f = floorf(py);
        float wx = px - x0f, wy = py - y0f;
        int x0 = (int)x0f, y0 = (int)y0f;
        float mv = lg[p] * sinv;

        float v = 0.f;
        #pragma unroll
        for (int cy = 0; cy < 2; ++cy) {
            int yi = y0 + cy;
            float wyv = cy ? wy : (1.f - wy);
            #pragma unroll
            for (int cx = 0; cx < 2; ++cx) {
                int xi = x0 + cx;
                float wgt = wyv * (cx ? wx : (1.f - wx));
                if (xi >= 0 && xi < HP && yi >= 0 && yi < HP)
                    v += img[((size_t)yi * HP + xi) * CCH] * wgt;
            }
        }
        acc += mv * v;
    }
    s16[(size_t)m * CCH + cidx] = (_Float16)acc;
}

// ---------------------------------------------------------------------------
extern "C" void kernel_launch(void* const* d_in, const int* in_sizes, int n_in,
                              void* d_out, int out_size, void* d_ws, size_t ws_size,
                              hipStream_t stream) {
    const float* x        = (const float*)d_in[0];
    const float* conv1_w  = (const float*)d_in[1];
    const float* bn1_g    = (const float*)d_in[2];
    const float* bn1_b    = (const float*)d_in[3];
    const float* bn1_m    = (const float*)d_in[4];
    const float* bn1_v    = (const float*)d_in[5];
    const float* inproj_w = (const float*)d_in[6];
    const float* inproj_b = (const float*)d_in[7];
    const float* dw_w     = (const float*)d_in[8];
    const float* dw_b     = (const float*)d_in[9];
    const float* ln_g     = (const float*)d_in[10];
    const float* ln_b     = (const float*)d_in[11];
    const float* off_w    = (const float*)d_in[12];
    const float* off_b    = (const float*)d_in[13];
    const float* mask_w   = (const float*)d_in[14];
    const float* mask_b   = (const float*)d_in[15];
    const float* outproj_w= (const float*)d_in[16];
    const float* outproj_b= (const float*)d_in[17];
    const float* bn2_g    = (const float*)d_in[18];
    const float* bn2_b    = (const float*)d_in[19];
    const float* bn2_m    = (const float*)d_in[20];
    const float* bn2_v    = (const float*)d_in[21];
    float* out = (float*)d_out;

    // workspace carve (256B aligned)
    size_t off = 0;
    char* base = (char*)d_ws;
    auto carve = [&](size_t bytes) -> char* {
        char* p = base + off; off += (bytes + 255) & ~(size_t)255; return p;
    };
    _Float16* a1    = (_Float16*)carve((size_t)MROWS * CCH * 2); // x NHWC f16, reused for z16
    float*    y     = (float*)   carve((size_t)MROWS * CCH * 4); // conv1+bn+silu NHWC f32
    _Float16* y16   = (_Float16*)carve((size_t)MROWS * CCH * 2); // f16 copy, reused for s16
    float*    xpad  = (float*)   carve((size_t)NB * HP * HP * CCH * 4);
    float*    omb   = (float*)   carve((size_t)MROWS * NOMP * 4);
    _Float16* wt_c1 = (_Float16*)carve((size_t)CCH * CCH * 2);
    _Float16* wt_in = (_Float16*)carve((size_t)CCH * CCH * 2);
    _Float16* wt_om = (_Float16*)carve((size_t)NOMP * CCH * 2);
    _Float16* wt_out= (_Float16*)carve((size_t)CCH * CCH * 2);
    float*    bias_om=(float*)   carve((size_t)NOMP * 4);
    _Float16* z16   = a1;        // reuse (a1 dead after GEMM1)
    _Float16* s16   = y16;       // reuse (y16 dead after GEMM2)

    // 1. weight pack/transpose to f16
    k_prep_weights<<<256, 256, 0, stream>>>(conv1_w, inproj_w, off_w, mask_w,
        outproj_w, off_b, mask_b, wt_c1, wt_in, wt_om, wt_out, bias_om);
    // 2. x NCHW -> NHWC f16
    k_x_to_nhwc16<<<(NB * CCH * HWD * HWD) / 256, 256, 0, stream>>>(x, a1);
    // 3. zero padded sampling image
    {
        int nz = NB * HP * HP * CCH;
        k_zero_f32<<<(nz + 255) / 256, 256, 0, stream>>>(xpad, nz);
    }
    // grid sizes: one wave per 32x32 tile
    const int blocks_256 = (MROWS / 32) * (CCH / 32) * 32 / 256;   // 256 blocks
    const int blocks_224 = (MROWS / 32) * (NOMP / 32) * 32 / 256;  // 224 blocks
    // 4. GEMM1: conv1 + BN1 + SiLU -> y (f32) + y16
    k_gemm<0><<<blocks_256, 256, 0, stream>>>(a1, wt_c1, CCH / 32, CCH,
        bn1_g, bn1_b, bn1_m, bn1_v, nullptr, y, y16);
    // 5. GEMM2: input_proj + bias -> padded NHWC image
    k_gemm<1><<<blocks_256, 256, 0, stream>>>(y16, wt_in, CCH / 32, CCH,
        inproj_b, nullptr, nullptr, nullptr, nullptr, xpad, nullptr);
    // 6. depthwise 3x3 + LN + GELU -> z16
    k_dw_ln_gelu<<<MROWS, 256, 0, stream>>>(y, dw_w, dw_b, ln_g, ln_b, z16);
    // 7. GEMM3: offset/mask logits
    k_gemm<2><<<blocks_224, 256, 0, stream>>>(z16, wt_om, NOMP / 32, CCH,
        bias_om, nullptr, nullptr, nullptr, nullptr, omb, nullptr);
    // 8. deformable bilinear sampling + fused softmax -> s16
    k_dcn<<<(MROWS * GRP * 32) / 256, 256, 0, stream>>>(xpad, omb, s16);
    // 9. GEMM4: output_proj + bias + BN2 + SiLU -> NCHW d_out
    k_gemm<3><<<blocks_256, 256, 0, stream>>>(s16, wt_out, CCH / 32, CCH,
        outproj_b, bn2_g, bn2_b, bn2_m, bn2_v, out, nullptr);
}
